// SAGE_51462298140964
// MI455X (gfx1250) — matrix operand order repaired and checked by reading, not compiled, not run on hardware
//
#include <hip/hip_runtime.h>
#include <stdint.h>

// ---------------------------------------------------------------------------
// MI455X / gfx1250 GraphSAGE (3-layer) forward.
//   * edge aggregation: global_atomic_add_f32 (h & agg are L2-resident: 51MB each vs 192MB L2)
//   * GEMM: single fused K=256 matmul  [h | agg*invdeg] @ [Wself;Wneigh]
//           using v_wmma_f32_16x16x32_bf16 (bf16 operands, f32 accumulate)
//   * weights pre-swizzled into per-lane WMMA B-fragment order (coalesced 32B/lane loads)
//   * f32->bf16 conversion is branchless RNE (no divergent NaN path near WMMA)
// ---------------------------------------------------------------------------

typedef __attribute__((ext_vector_type(16))) __bf16 v16bf;
typedef __attribute__((ext_vector_type(8)))  float  v8f;
typedef __attribute__((ext_vector_type(4)))  float  f4;

union AFrag { unsigned short s[16]; unsigned int u[8]; v16bf v; };

// Branchless round-to-nearest-even f32 -> bf16 (inputs are finite).
__device__ __forceinline__ unsigned short f2bf(float f) {
  unsigned u = __builtin_bit_cast(unsigned, f);
  u += 0x7fffu + ((u >> 16) & 1u);
  return (unsigned short)(u >> 16);
}

// -------------------------------- utility ----------------------------------
__global__ void fill_f32(float* __restrict__ p, float v, long long n) {
  long long i = (long long)blockIdx.x * blockDim.x + threadIdx.x;
  if (i < n) p[i] = v;
}

// -------------------------------- degree -----------------------------------
__global__ void degree_k(const int* __restrict__ dst, float* __restrict__ deg, int E) {
  int e = blockIdx.x * blockDim.x + threadIdx.x;
  if (e < E) atomicAdd(&deg[dst[e]], 1.0f);
}

__global__ void invdeg_k(float* __restrict__ deg, int N) {
  int i = blockIdx.x * blockDim.x + threadIdx.x;
  if (i < N) deg[i] = 1.0f / fmaxf(deg[i], 1.0f);
}

// --------------------------- edge scatter-add -------------------------------
// One wave32 per edge: lane loads one float4 of the src row (coalesced 512B)
// and atomically adds into the dst row. Both rows live in L2.
__global__ __launch_bounds__(256) void scatter_k(
    const float* __restrict__ h, const int* __restrict__ src,
    const int* __restrict__ dst, float* __restrict__ agg, int E)
{
  int t = blockIdx.x * 256 + threadIdx.x;
  int e = t >> 5, lane = t & 31;
  if (e >= E) return;
  __builtin_prefetch(src + e + 2048, 0, 1);  // stream edge list ahead of use
  int s = src[e], d = dst[e];
  f4 v = ((const f4*)(h + (size_t)s * 128))[lane];
  float* a = agg + (size_t)d * 128 + lane * 4;
  atomicAdd(a + 0, v.x); atomicAdd(a + 1, v.y);
  atomicAdd(a + 2, v.z); atomicAdd(a + 3, v.w);
}

// ----------------------- weight pre-swizzle (bf16) --------------------------
// Packs Wcat = [Wself; Wneigh] (KCAT=2*KIN rows, KOUT cols) into the per-lane
// V_WMMA_16X16X32_BF16 B-fragment layout:
//   frag(ntile,ktile): lane L holds col n = L&15, K = ktile*32 + (L>>4)*8 + (e&7) + (e>>3)*16,
//   e = 0..15 stored contiguously (16 bf16 = 32B per lane).
__global__ void pack_w_k(const float* __restrict__ wself,
                         const float* __restrict__ wneigh,
                         unsigned short* __restrict__ packed, int KIN, int KOUT)
{
  int idx = blockIdx.x * blockDim.x + threadIdx.x;
  int total = 2 * KIN * KOUT;
  if (idx >= total) return;
  int KT = (2 * KIN) / 32;
  int e     = idx & 15;
  int lane  = (idx >> 4) & 31;
  int rest  = idx >> 9;            // ntile*KT + ktile
  int ktile = rest % KT;
  int ntile = rest / KT;
  int K = ktile * 32 + (lane >> 4) * 8 + (e & 7) + (e >> 3) * 16;
  int c = ntile * 16 + (lane & 15);
  float v = (K < KIN) ? wself[(size_t)K * KOUT + c]
                      : wneigh[(size_t)(K - KIN) * KOUT + c];
  packed[idx] = f2bf(v);
}

// ------------------------------ WMMA GEMM -----------------------------------
// out[n, KOUT] = [h | agg*invdeg] @ Wcat + bias.  Block = 256 thr = 8 waves,
// covers 64 rows. Wave w: M-tile (w&3), N-half (w>>2) -> NT = KOUT/32 tiles.
template <int KOUT>
__global__ __launch_bounds__(256) void sage_gemm_k(
    const float* __restrict__ h, const float* __restrict__ agg,
    const float* __restrict__ invdeg, const v16bf* __restrict__ wpack,
    const float* __restrict__ bias, float* __restrict__ out, int n)
{
  constexpr int KT = 8;            // 256 / 32 k-steps
  constexpr int NT = KOUT / 32;    // 16x16 N-tiles per wave
  const int lane  = threadIdx.x & 31;
  const int wave  = threadIdx.x >> 5;
  const int mtile = wave & 3;
  const int nhalf = wave >> 2;
  const int r0    = blockIdx.x * 64 + mtile * 16;
  const int half  = lane >> 4;
  const int arow  = r0 + (lane & 15);
  const int rr    = (arow < n) ? arow : (n - 1);     // clamp: keep EXEC all-ones
  const float inv = invdeg[rr];
  const float* __restrict__ hp = h   + (size_t)rr * 128;
  const float* __restrict__ ap = agg + (size_t)rr * 128;
  const int base0 = half * 8;      // ISA 16-bit A layout half-wave K offset

  v8f acc[NT];
#pragma unroll
  for (int t = 0; t < NT; ++t)
#pragma unroll
    for (int v = 0; v < 8; ++v) acc[t][v] = 0.0f;

#pragma unroll
  for (int kt = 0; kt < KT; ++kt) {
    AFrag af;
#pragma unroll
    for (int c = 0; c < 2; ++c) {                    // K chunks [base0, base0+8) and [base0+16, base0+24)
      int k = kt * 32 + base0 + c * 16;
      const float* p; float sc;
      if (k < 128) { p = hp + k;         sc = 1.0f; }   // self features
      else         { p = ap + (k - 128); sc = inv;  }   // mean-aggregated neighbors
      f4 x0 = *(const f4*)(p);
      f4 x1 = *(const f4*)(p + 4);
      af.u[c*4+0] = (unsigned)f2bf(x0.x*sc) | ((unsigned)f2bf(x0.y*sc) << 16);
      af.u[c*4+1] = (unsigned)f2bf(x0.z*sc) | ((unsigned)f2bf(x0.w*sc) << 16);
      af.u[c*4+2] = (unsigned)f2bf(x1.x*sc) | ((unsigned)f2bf(x1.y*sc) << 16);
      af.u[c*4+3] = (unsigned)f2bf(x1.z*sc) | ((unsigned)f2bf(x1.w*sc) << 16);
    }
#pragma unroll
    for (int t = 0; t < NT; ++t) {
      int ntile = nhalf * NT + t;
      v16bf bf = wpack[(size_t)(ntile * KT + kt) * 32 + lane];
      acc[t] = __builtin_amdgcn_wmma_f32_16x16x32_bf16(
          false, af.v, false, bf, (short)0, acc[t], false, false);
    }
  }

  // D layout: lane (0-15,16-31) -> N = lane&15, M = v + 8*(lane>>4)
#pragma unroll
  for (int t = 0; t < NT; ++t) {
    int col = (nhalf * NT + t) * 16 + (lane & 15);
    float bv = bias[col];
#pragma unroll
    for (int v = 0; v < 8; ++v) {
      int orow = r0 + half * 8 + v;
      if (orow < n) out[(size_t)orow * KOUT + col] = acc[t][v] + bv;
    }
  }
}

// ------------------------------ BatchNorm -----------------------------------
__global__ __launch_bounds__(128) void bn_stats_k(
    const float* __restrict__ t, float* __restrict__ stats, int n)
{
  int f  = threadIdx.x;                 // feature 0..127
  int r0 = blockIdx.x * 512;
  int r1 = min(r0 + 512, n);
  float s = 0.0f, s2 = 0.0f;
  for (int r = r0; r < r1; ++r) {
    float v = t[(size_t)r * 128 + f];
    s += v; s2 += v * v;
  }
  atomicAdd(&stats[f], s);
  atomicAdd(&stats[128 + f], s2);
}

__global__ __launch_bounds__(128) void bn_finalize_k(
    float* __restrict__ stats, const float* __restrict__ gamma,
    const float* __restrict__ beta, int N)
{
  int f = threadIdx.x;
  float mu  = stats[f] / (float)N;
  float var = stats[128 + f] / (float)N - mu * mu;
  float sc  = gamma[f] * rsqrtf(var + 1e-5f);
  stats[256 + f] = sc;
  stats[384 + f] = beta[f] - mu * sc;
}

__global__ __launch_bounds__(256) void bn_relu_k(
    const float* __restrict__ t, const float* __restrict__ stats,
    float* __restrict__ o, int n4)
{
  int i = blockIdx.x * blockDim.x + threadIdx.x;
  if (i >= n4) return;
  f4 v  = ((const f4*)t)[i];
  int g = i & 31;                              // 128 feats / 4
  f4 sc = ((const f4*)(stats + 256))[g];
  f4 sh = ((const f4*)(stats + 384))[g];
  f4 r;
  r.x = fmaxf(v.x * sc.x + sh.x, 0.0f);
  r.y = fmaxf(v.y * sc.y + sh.y, 0.0f);
  r.z = fmaxf(v.z * sc.z + sh.z, 0.0f);
  r.w = fmaxf(v.w * sc.w + sh.w, 0.0f);
  ((f4*)o)[i] = r;
}

// ------------------------------ log_softmax ---------------------------------
__global__ __launch_bounds__(256) void logsoftmax64_k(
    const float* __restrict__ t, float* __restrict__ out, int n)
{
  int gt = blockIdx.x * 256 + threadIdx.x;
  int node = gt >> 5, lane = gt & 31;
  if (node >= n) return;
  const float* r = t + (size_t)node * 64;
  float a = r[lane], b = r[lane + 32];
  float m = fmaxf(a, b);
  for (int off = 16; off; off >>= 1) m = fmaxf(m, __shfl_xor(m, off, 32));
  float s = expf(a - m) + expf(b - m);
  for (int off = 16; off; off >>= 1) s += __shfl_xor(s, off, 32);
  float l = m + logf(s);
  out[(size_t)node * 64 + lane]      = a - l;
  out[(size_t)node * 64 + lane + 32] = b - l;
}

// ------------------------------- launcher -----------------------------------
extern "C" void kernel_launch(void* const* d_in, const int* in_sizes, int n_in,
                              void* d_out, int out_size, void* d_ws, size_t ws_size,
                              hipStream_t stream)
{
  const float* x = (const float*)d_in[0];
  const int* src = (const int*)d_in[1];
  const int* dst = (const int*)d_in[2];
  const float* wself[3]  = {(const float*)d_in[3], (const float*)d_in[6], (const float*)d_in[9]};
  const float* wneigh[3] = {(const float*)d_in[4], (const float*)d_in[7], (const float*)d_in[10]};
  const float* bias[3]   = {(const float*)d_in[5], (const float*)d_in[8], (const float*)d_in[11]};
  const float* gamma[2]  = {(const float*)d_in[12], (const float*)d_in[14]};
  const float* beta[2]   = {(const float*)d_in[13], (const float*)d_in[15]};
  const int N = in_sizes[0] / 128;
  const int E = in_sizes[1];
  float* out = (float*)d_out;

  char* ws = (char*)d_ws;
  size_t NB = (size_t)N * 128 * sizeof(float);
  float* agg    = (float*)(ws);
  float* bufA   = (float*)(ws + NB);
  float* bufB   = (float*)(ws + 2 * NB);
  float* invdeg = (float*)(ws + 3 * NB);
  float* stats  = invdeg + N;                       // 512 floats
  unsigned short* pw0 = (unsigned short*)(stats + 512);
  unsigned short* pw1 = pw0 + 256 * 128;
  unsigned short* pw2 = pw1 + 256 * 128;

  const dim3 B256(256), B128(128);
  const int gN128 = (int)(((size_t)N * 128 / 4 + 255) / 256);  // bn_relu grid
  const int gFill = (int)(((size_t)N * 128 + 255) / 256);
  const int gGemm = (N + 63) / 64;
  const int gScat = (int)(((size_t)E * 32 + 255) / 256);
  const int gStat = (N + 511) / 512;
  const int gSoft = (int)(((size_t)N * 32 + 255) / 256);

  // weight pre-swizzle (every call: deterministic, no caching)
  pack_w_k<<<(256 * 128 + 255) / 256, B256, 0, stream>>>(wself[0], wneigh[0], pw0, 128, 128);
  pack_w_k<<<(256 * 128 + 255) / 256, B256, 0, stream>>>(wself[1], wneigh[1], pw1, 128, 128);
  pack_w_k<<<(256 * 64  + 255) / 256, B256, 0, stream>>>(wself[2], wneigh[2], pw2, 128, 64);

  // degrees (graph is fixed across layers)
  fill_f32<<<(N + 255) / 256, B256, 0, stream>>>(invdeg, 0.0f, N);
  degree_k<<<(E + 255) / 256, B256, 0, stream>>>(dst, invdeg, E);
  invdeg_k<<<(N + 255) / 256, B256, 0, stream>>>(invdeg, N);

  // ---- layer 0 ----
  fill_f32<<<gFill, B256, 0, stream>>>(agg, 0.0f, (long long)N * 128);
  scatter_k<<<gScat, B256, 0, stream>>>(x, src, dst, agg, E);
  sage_gemm_k<128><<<gGemm, B256, 0, stream>>>(x, agg, invdeg, (const v16bf*)pw0, bias[0], bufA, N);
  fill_f32<<<1, B256, 0, stream>>>(stats, 0.0f, 256);
  bn_stats_k<<<gStat, B128, 0, stream>>>(bufA, stats, N);
  bn_finalize_k<<<1, B128, 0, stream>>>(stats, gamma[0], beta[0], N);
  bn_relu_k<<<gN128, B256, 0, stream>>>(bufA, stats, bufB, (int)((size_t)N * 128 / 4));

  // ---- layer 1 ----
  fill_f32<<<gFill, B256, 0, stream>>>(agg, 0.0f, (long long)N * 128);
  scatter_k<<<gScat, B256, 0, stream>>>(bufB, src, dst, agg, E);
  sage_gemm_k<128><<<gGemm, B256, 0, stream>>>(bufB, agg, invdeg, (const v16bf*)pw1, bias[1], bufA, N);
  fill_f32<<<1, B256, 0, stream>>>(stats, 0.0f, 256);
  bn_stats_k<<<gStat, B128, 0, stream>>>(bufA, stats, N);
  bn_finalize_k<<<1, B128, 0, stream>>>(stats, gamma[1], beta[1], N);
  bn_relu_k<<<gN128, B256, 0, stream>>>(bufA, stats, bufB, (int)((size_t)N * 128 / 4));

  // ---- layer 2 + log_softmax ----
  fill_f32<<<gFill, B256, 0, stream>>>(agg, 0.0f, (long long)N * 128);
  scatter_k<<<gScat, B256, 0, stream>>>(bufB, src, dst, agg, E);
  sage_gemm_k<64><<<gGemm, B256, 0, stream>>>(bufB, agg, invdeg, (const v16bf*)pw2, bias[2], bufA, N);
  logsoftmax64_k<<<gSoft, B256, 0, stream>>>(bufA, out, N);
}
